// PointConditioner_72335839199727
// MI455X (gfx1250) — matrix operand order repaired and checked by reading, not compiled, not run on hardware
//
#include <hip/hip_runtime.h>
#include <math.h>

typedef _Float16 half8 __attribute__((ext_vector_type(8)));
typedef _Float16 v16h  __attribute__((ext_vector_type(16)));
typedef float    v8f   __attribute__((ext_vector_type(8)));

#define CFPN  128
#define CC    64
#define BATCH 8
#define NPTS  8192
#define WSTR  136   // padded half-stride for W rows in LDS (avoids 64-bank conflicts)
#define PSTR  136   // padded half-stride for P-tile rows in LDS
#define OSTR  68    // padded float-stride for output staging (same bytes as PSTR halves)

// ---------------------------------------------------------------------------
// Phase 1: 1x1 conv as WMMA GEMM.  Q[b][pix][o] = sum_c W[o][c]*P[b][c][pix] + bias[o]
// Q stored channels-last so phase-2 gathers read contiguous 256B vectors (L2-resident).
// Block: 256 threads (8 waves). Each block: 128 pixels. Each wave: 16 pixels x 64 outs.
// ---------------------------------------------------------------------------
__global__ __launch_bounds__(256) void conv1x1_wmma(const float* __restrict__ P,
                                                    const float* __restrict__ Wg,
                                                    const float* __restrict__ bg,
                                                    float* __restrict__ Q,
                                                    int HW) {
    __shared__ _Float16 sW[CC * WSTR];       // 17 KB
    __shared__ float    sBias[CC];
    __shared__ _Float16 sP[128 * PSTR];      // 34 KB (reused as f32 out staging)

    const int tid        = threadIdx.x;
    const int tilesPerB  = HW >> 7;
    const int b          = blockIdx.x / tilesPerB;
    const int pixBase    = (blockIdx.x % tilesPerB) << 7;

    // Stage W (64x128) as f16, coalesced read.
    for (int idx = tid; idx < CC * CFPN; idx += 256) {
        int o = idx >> 7, c = idx & 127;
        sW[o * WSTR + c] = (_Float16)Wg[idx];
    }
    if (tid < CC) sBias[tid] = bg[tid];

    // Stage P tile (128 channels x 128 pixels), transposed to [pix][c] f16.
    const float* Pb = P + (size_t)b * CFPN * HW + pixBase;
    for (int idx = tid; idx < CFPN * 128; idx += 256) {
        int c = idx >> 7, pix = idx & 127;               // consecutive tid -> consecutive pix (coalesced)
        sP[pix * PSTR + c] = (_Float16)Pb[(size_t)c * HW + pix];
    }
    __syncthreads();

    const int wave = tid >> 5;
    const int lane = tid & 31;
    const int m    = lane & 15;      // N-column (pixel) / A-row (out channel) lane index
    const int h    = lane >> 4;      // half-wave select
    const int pix0 = wave << 4;

    v8f acc[4];
    #pragma unroll
    for (int t = 0; t < 4; ++t)
        #pragma unroll
        for (int v = 0; v < 8; ++v)
            acc[t][v] = sBias[t * 16 + v + h * 8];       // C/D layout: M = v + h*8

    const _Float16* prow = &sP[(pix0 + m) * PSTR];

    #pragma unroll
    for (int ks = 0; ks < CFPN; ks += 32) {
        // B fragment: 32x16, lane(h,m) holds K = {h*8+0..7, 16+h*8+0..7} for column n=m.
        half8 b0 = *(const half8*)(prow + ks + h * 8);
        half8 b1 = *(const half8*)(prow + ks + 16 + h * 8);
        v16h bf  = __builtin_shufflevector(b0, b1, 0,1,2,3,4,5,6,7,8,9,10,11,12,13,14,15);
        #pragma unroll
        for (int t = 0; t < 4; ++t) {
            // A fragment: 16x32, lane(h,m) row o = t*16+m, K = {h*8+0..7, 16+h*8+0..7}.
            const _Float16* wr = &sW[(t * 16 + m) * WSTR + ks];
            half8 a0 = *(const half8*)(wr + h * 8);
            half8 a1 = *(const half8*)(wr + 16 + h * 8);
            v16h af  = __builtin_shufflevector(a0, a1, 0,1,2,3,4,5,6,7,8,9,10,11,12,13,14,15);
            acc[t] = __builtin_amdgcn_wmma_f32_16x16x32_f16(false, af, false, bf,
                                                            (short)0, acc[t], false, false);
        }
    }

    // Transpose through LDS (reuse sP region) so global stores are coalesced float4s.
    __syncthreads();
    float* sOut = (float*)sP;        // 128 rows x OSTR floats
    #pragma unroll
    for (int t = 0; t < 4; ++t)
        #pragma unroll
        for (int v = 0; v < 8; ++v)
            sOut[(pix0 + m) * OSTR + t * 16 + v + h * 8] = acc[t][v];
    __syncthreads();

    float* Qb = Q + ((size_t)b * HW + pixBase) * CC;
    for (int idx = tid; idx < 128 * 16; idx += 256) {
        int pix = idx >> 4, sub = idx & 15;
        *(float4*)(Qb + (size_t)pix * CC + sub * 4) =
            *(const float4*)(sOut + pix * OSTR + sub * 4);
    }
}

// ---------------------------------------------------------------------------
// Phase 2: bilinear sample (zeros padding, align_corners=False) from channels-last Q.
// One wave per point; each lane handles 2 channels -> 256B coalesced loads/stores.
// Q (44 MB total) is L2-resident, so the 4-corner gathers mostly hit L2.
// ---------------------------------------------------------------------------
__global__ __launch_bounds__(256) void bilinear_sample_kernel(const float* __restrict__ Q4,
                                                              const float* __restrict__ Q8,
                                                              const float* __restrict__ Q16,
                                                              const float* __restrict__ pn,
                                                              float* __restrict__ out) {
    const int tid  = threadIdx.x;
    const int lane = tid & 31;
    const int g    = blockIdx.x * 8 + (tid >> 5);  // global point id
    const int b    = g / NPTS;
    const int n    = g % NPTS;

    const float px = pn[((size_t)b * NPTS + n) * 2 + 0];
    const float py = pn[((size_t)b * NPTS + n) * 2 + 1];
    float* orow = out + ((size_t)b * NPTS + n) * 192 + lane * 2;

    const float* Qs[3] = {Q4, Q8, Q16};
    const int    Ss[3] = {128, 64, 32};

    #pragma unroll
    for (int l = 0; l < 3; ++l) {
        const int S = Ss[l];
        const float* Q = Qs[l];
        float x = (px + 1.0f) * (S * 0.5f) - 0.5f;
        float y = (py + 1.0f) * (S * 0.5f) - 0.5f;
        float x0f = floorf(x), y0f = floorf(y);
        float wx1 = x - x0f, wx0 = 1.0f - wx1;
        float wy1 = y - y0f, wy0 = 1.0f - wy1;
        int x0 = (int)x0f, y0 = (int)y0f;

        float2 acc = make_float2(0.0f, 0.0f);
        #pragma unroll
        for (int cy = 0; cy < 2; ++cy) {
            #pragma unroll
            for (int cx = 0; cx < 2; ++cx) {
                int xi = x0 + cx, yi = y0 + cy;
                bool valid = (xi >= 0) && (xi < S) && (yi >= 0) && (yi < S);
                int xc = min(max(xi, 0), S - 1);
                int yc = min(max(yi, 0), S - 1);
                float w = (cx ? wx1 : wx0) * (cy ? wy1 : wy0);
                const float2 v = *(const float2*)(Q + (((size_t)b * S * S) + (size_t)yc * S + xc) * CC
                                                  + lane * 2);
                if (valid) { acc.x += w * v.x; acc.y += w * v.y; }
            }
        }
        *(float2*)(orow + l * 64) = acc;
    }
}

// ---------------------------------------------------------------------------
extern "C" void kernel_launch(void* const* d_in, const int* in_sizes, int n_in,
                              void* d_out, int out_size, void* d_ws, size_t ws_size,
                              hipStream_t stream) {
    (void)in_sizes; (void)n_in; (void)out_size; (void)ws_size;
    const float* P4   = (const float*)d_in[0];
    const float* P8   = (const float*)d_in[1];
    const float* P16  = (const float*)d_in[2];
    const float* pn   = (const float*)d_in[3];
    const float* W4   = (const float*)d_in[4];
    const float* b4   = (const float*)d_in[5];
    const float* W8   = (const float*)d_in[6];
    const float* b8   = (const float*)d_in[7];
    const float* W16  = (const float*)d_in[8];
    const float* b16  = (const float*)d_in[9];
    float* out = (float*)d_out;

    // Workspace: channels-last conditioned maps (44 MB total, L2-resident)
    float* q4  = (float*)d_ws;
    float* q8  = q4 + (size_t)BATCH * 16384 * CC;
    float* q16 = q8 + (size_t)BATCH * 4096 * CC;

    conv1x1_wmma<<<BATCH * (16384 / 128), 256, 0, stream>>>(P4,  W4,  b4,  q4,  16384);
    conv1x1_wmma<<<BATCH * (4096  / 128), 256, 0, stream>>>(P8,  W8,  b8,  q8,  4096);
    conv1x1_wmma<<<BATCH * (1024  / 128), 256, 0, stream>>>(P16, W16, b16, q16, 1024);
    bilinear_sample_kernel<<<(BATCH * NPTS) / 8, 256, 0, stream>>>(q4, q8, q16, pn, out);
}
